// Net_88029649699335
// MI455X (gfx1250) — compile-verified, matrix operand order checked by reading, not run on hardware
//
#include <hip/hip_runtime.h>
#include <hip/hip_bf16.h>

// ---------------- problem constants (match reference) ----------------
static constexpr int N_NODES = 50000;
static constexpr int N_EDGES = 640000;
static constexpr int N_EDGES_SL = N_EDGES + N_NODES;   // with self loops
static constexpr int G_GRAPHS = 128;
static constexpr int IN_DIM = 128;
static constexpr int LAT = 32;
static constexpr int H_HEADS = 8;
static constexpr int HL = H_HEADS * LAT;               // 256
static constexpr int HID = 128;
static constexpr int NC = 2;
static constexpr int CAT = LAT + 2 * HL;               // 544

typedef __attribute__((ext_vector_type(16))) _Float16 v16h;
typedef __attribute__((ext_vector_type(8)))  float    v8f;

// ---------------- helpers ----------------
__device__ __forceinline__ float elu1(float x) {
  return x > 0.0f ? x : (__expf(x) - 1.0f);
}
__device__ __forceinline__ float lrelu02(float x) {
  return x > 0.0f ? x : 0.2f * x;
}
__device__ __forceinline__ void edge_rc(const int* __restrict__ row,
                                        const int* __restrict__ col,
                                        int e, int& r, int& c) {
  if (e < N_EDGES) { r = row[e]; c = col[e]; }
  else             { r = e - N_EDGES; c = r; }   // self loop
}

// ---------------- generic fill ----------------
__global__ void fill_kernel(float* __restrict__ p, float v, long long n) {
  long long t = (long long)blockIdx.x * blockDim.x + threadIdx.x;
  if (t < n) p[t] = v;
}

// ---------------- weight pre-pack: f32 [K,Nc] -> f16 WMMA B-fragment order ----
// Fragment layout: Bp[(((tn*(K/32) + kb)*32 + lane)*16 + c]
//   holds B[k, n] with k = (c&7) + 8*(lane>>4) + 16*(c>>3) + 32*kb, n = tn*16 + (lane&15).
// Each lane of a consuming wave then reads its 16 halfs as one aligned 32B vector.
__global__ void pack_b_kernel(const float* __restrict__ B, _Float16* __restrict__ Bp,
                              int K, int Nc) {
  int t = blockIdx.x * blockDim.x + threadIdx.x;
  if (t >= K * Nc) return;
  int c    = t & 15;
  int lane = (t >> 4) & 31;
  int rest = t >> 9;
  int kblocks = K >> 5;
  int kb = rest % kblocks;
  int tn = rest / kblocks;
  int k = (c & 7) + 8 * (lane >> 4) + 16 * (c >> 3) + (kb << 5);
  int n = (tn << 4) + (lane & 15);
  Bp[t] = (_Float16)B[(size_t)k * Nc + n];
}

// ---------------- WMMA GEMM: C[M,Nc] = A[M,K] @ Bpack (+bias) ----------------
// One wave32 computes a 2x2 block of 16x16 tiles (32x32 outputs): A reused over
// 2 N-tiles, B reused over 2 M-tiles, 4 v_wmma per 32-deep k step.
// A: f32 row-major (rows loaded as 4x float4 -> cvt_pk to f16).
// B: pre-packed f16 fragments (one 32B vector load per lane per tile).
__device__ __forceinline__ v16h load_a_frag(const float* __restrict__ Arow, int lg) {
  const float4* p = reinterpret_cast<const float4*>(Arow);
  float4 q0 = p[2 * lg + 0];
  float4 q1 = p[2 * lg + 1];
  float4 q2 = p[2 * lg + 4];
  float4 q3 = p[2 * lg + 5];
  v16h a;
  a[0]  = (_Float16)q0.x; a[1]  = (_Float16)q0.y; a[2]  = (_Float16)q0.z; a[3]  = (_Float16)q0.w;
  a[4]  = (_Float16)q1.x; a[5]  = (_Float16)q1.y; a[6]  = (_Float16)q1.z; a[7]  = (_Float16)q1.w;
  a[8]  = (_Float16)q2.x; a[9]  = (_Float16)q2.y; a[10] = (_Float16)q2.z; a[11] = (_Float16)q2.w;
  a[12] = (_Float16)q3.x; a[13] = (_Float16)q3.y; a[14] = (_Float16)q3.z; a[15] = (_Float16)q3.w;
  return a;
}

__device__ __forceinline__ v16h load_b_frag(const _Float16* __restrict__ Bp,
                                            int tn, int kb, int kblocks, int lane) {
  const v16h* p = reinterpret_cast<const v16h*>(
      Bp + ((((size_t)tn * kblocks) + kb) * 32 + lane) * 16);
  return *p;
}

template <bool HAS_BIAS>
__global__ void gemm_wmma_kernel(const float* __restrict__ A,
                                 const _Float16* __restrict__ Bp,
                                 float* __restrict__ C,
                                 int M, int K, int Nc,
                                 const float* __restrict__ bias) {
  const int tilesM = M >> 4, tilesN = Nc >> 4;
  const int tilesM2 = (tilesM + 1) >> 1, tilesN2 = (tilesN + 1) >> 1;
  const int wave = blockIdx.x * (blockDim.x >> 5) + (threadIdx.x >> 5);
  if (wave >= tilesM2 * tilesN2) return;   // whole-wave guard: EXEC stays all-ones

  const int tm0 = (wave / tilesN2) * 2;
  const int tn0 = (wave % tilesN2) * 2;
  const int tm1 = min(tm0 + 1, tilesM - 1);   // clamp: duplicate tile on odd edge (benign)
  const int tn1 = min(tn0 + 1, tilesN - 1);

  const int lane = threadIdx.x & 31;
  const int lg   = lane >> 4;
  const int m16  = lane & 15;
  const int kblocks = K >> 5;

  const float* Arow0 = A + (size_t)(tm0 * 16 + m16) * K;
  const float* Arow1 = A + (size_t)(tm1 * 16 + m16) * K;

  v8f acc00 = {}, acc01 = {}, acc10 = {}, acc11 = {};
  for (int kb = 0; kb < kblocks; ++kb) {
    v16h a0 = load_a_frag(Arow0 + (kb << 5), lg);
    v16h a1 = load_a_frag(Arow1 + (kb << 5), lg);
    v16h b0 = load_b_frag(Bp, tn0, kb, kblocks, lane);
    v16h b1 = load_b_frag(Bp, tn1, kb, kblocks, lane);
    acc00 = __builtin_amdgcn_wmma_f32_16x16x32_f16(false, a0, false, b0, (short)0, acc00, false, false);
    acc01 = __builtin_amdgcn_wmma_f32_16x16x32_f16(false, a0, false, b1, (short)0, acc01, false, false);
    acc10 = __builtin_amdgcn_wmma_f32_16x16x32_f16(false, a1, false, b0, (short)0, acc10, false, false);
    acc11 = __builtin_amdgcn_wmma_f32_16x16x32_f16(false, a1, false, b1, (short)0, acc11, false, false);
  }

  const int n0 = tn0 * 16 + m16;
  const int n1 = tn1 * 16 + m16;
  float bias0 = 0.0f, bias1 = 0.0f;
  if (HAS_BIAS) { bias0 = bias[n0]; bias1 = bias[n1]; }

  // D layout: VGPR r holds row (r + 8*lg), column = lane&15
#pragma unroll
  for (int r = 0; r < 8; ++r) {
    const int m0 = tm0 * 16 + r + 8 * lg;
    const int m1 = tm1 * 16 + r + 8 * lg;
    C[(size_t)m0 * Nc + n0] = acc00[r] + bias0;
    C[(size_t)m0 * Nc + n1] = acc01[r] + bias1;
    C[(size_t)m1 * Nc + n0] = acc10[r] + bias0;
    C[(size_t)m1 * Nc + n1] = acc11[r] + bias1;
  }
}

// ---------------- GCN ----------------
__global__ void gcn_deg_kernel(const int* __restrict__ col, float* __restrict__ deg) {
  int e = blockIdx.x * blockDim.x + threadIdx.x;
  if (e < N_EDGES) atomicAdd(&deg[col[e]], 1.0f);
}

__global__ void gcn_scatter_kernel(const int* __restrict__ row, const int* __restrict__ col,
                                   const float* __restrict__ deg,
                                   const float* __restrict__ hW,   // [N, LAT]
                                   float* __restrict__ acc) {      // [N, LAT]
  long long t = (long long)blockIdx.x * blockDim.x + threadIdx.x;
  if (t >= (long long)N_EDGES_SL * LAT) return;
  int e  = (int)(t >> 5);
  int ch = (int)(t & 31);
  int r, c; edge_rc(row, col, e, r, c);
  float norm = __frsqrt_rn(deg[r]) * __frsqrt_rn(deg[c]);   // deg >= 1 (self loops)
  atomicAdd(&acc[(size_t)c * LAT + ch], norm * hW[(size_t)r * LAT + ch]);
}

__global__ void bias_elu_kernel(const float* __restrict__ acc,
                                const float* __restrict__ bias,
                                float* __restrict__ out, int C, long long n) {
  long long t = (long long)blockIdx.x * blockDim.x + threadIdx.x;
  if (t >= n) return;
  int ch = (int)(t % C);
  out[t] = elu1(acc[t] + bias[ch]);
}

// ---------------- GAT ----------------
// s[n,h] = <hW[n,h,:], asrc[h,:]>, d[n,h] = <hW[n,h,:], adst[h,:]>
__global__ void gat_sd_kernel(const float* __restrict__ hW,
                              const float* __restrict__ asrc,
                              const float* __restrict__ adst,
                              float* __restrict__ s, float* __restrict__ d) {
  int t = blockIdx.x * blockDim.x + threadIdx.x;
  if (t >= N_NODES * H_HEADS) return;
  int n = t >> 3, h = t & 7;
  const float* hp = hW + (size_t)n * HL + h * LAT;
  const float* ap = asrc + h * LAT;
  const float* bp = adst + h * LAT;
  float sv = 0.f, dv = 0.f;
#pragma unroll
  for (int c = 0; c < LAT; ++c) { sv += hp[c] * ap[c]; dv += hp[c] * bp[c]; }
  s[t] = sv; d[t] = dv;
}

// z[c,h] += exp(leaky(s[r,h]+d[c,h]))   (max-shift dropped: cancels in alpha)
__global__ void gat_z_kernel(const int* __restrict__ row, const int* __restrict__ col,
                             const float* __restrict__ s, const float* __restrict__ d,
                             float* __restrict__ z) {
  long long t = (long long)blockIdx.x * blockDim.x + threadIdx.x;
  if (t >= (long long)N_EDGES_SL * H_HEADS) return;
  int e = (int)(t >> 3), h = (int)(t & 7);
  int r, c; edge_rc(row, col, e, r, c);
  float w = __expf(lrelu02(s[r * H_HEADS + h] + d[c * H_HEADS + h]));
  atomicAdd(&z[c * H_HEADS + h], w);
}

// acc[c, j] += alpha(e,h) * hW[r, j]   with j in [0,256), h = j>>5
__global__ void gat_scatter_kernel(const int* __restrict__ row, const int* __restrict__ col,
                                   const float* __restrict__ s, const float* __restrict__ d,
                                   const float* __restrict__ z,
                                   const float* __restrict__ hW,
                                   float* __restrict__ acc) {
  long long t = (long long)blockIdx.x * blockDim.x + threadIdx.x;
  if (t >= (long long)N_EDGES_SL * HL) return;
  int e = (int)(t >> 8);
  int j = (int)(t & 255);
  int h = j >> 5;
  int r, c; edge_rc(row, col, e, r, c);
  float w = __expf(lrelu02(s[r * H_HEADS + h] + d[c * H_HEADS + h]));
  float alpha = w / (z[c * H_HEADS + h] + 1e-16f);
  atomicAdd(&acc[(size_t)c * HL + j], alpha * hW[(size_t)r * HL + j]);
}

// ---------------- readout ----------------
__global__ void idx_kernel(const int* __restrict__ batch, int* __restrict__ idx) {
  int g = blockIdx.x * blockDim.x + threadIdx.x;
  if (g >= G_GRAPHS) return;
  int lo = 0, hi = N_NODES;                       // searchsorted(batch, g) left
  while (lo < hi) { int mid = (lo + hi) >> 1; if (batch[mid] < g) lo = mid + 1; else hi = mid; }
  idx[g] = lo;
}

__global__ void gather_cat_kernel(const int* __restrict__ idx,
                                  const float* __restrict__ h1,
                                  const float* __restrict__ h2,
                                  const float* __restrict__ h3,
                                  float* __restrict__ catG) {
  int t = blockIdx.x * blockDim.x + threadIdx.x;
  if (t >= G_GRAPHS * CAT) return;
  int g = t / CAT, j = t % CAT;
  int i = idx[g];
  float v;
  if (j < LAT)            v = h1[(size_t)i * LAT + j];
  else if (j < LAT + HL)  v = h2[(size_t)i * HL + (j - LAT)];
  else                    v = h3[(size_t)i * HL + (j - LAT - HL)];
  catG[t] = v;
}

__global__ void hidden_kernel(const float* __restrict__ feature, float* __restrict__ hid) {
  int t = blockIdx.x * blockDim.x + threadIdx.x;
  if (t < G_GRAPHS * HID) hid[t] = elu1(feature[t]);
}

__global__ void raw_kernel(const float* __restrict__ hid,
                           const float* __restrict__ Wl2,
                           const float* __restrict__ bl2,
                           float* __restrict__ raw) {
  int t = blockIdx.x * blockDim.x + threadIdx.x;
  if (t >= G_GRAPHS * NC) return;
  int g = t >> 1, c = t & 1;
  float acc = bl2[c];
#pragma unroll 16
  for (int k = 0; k < HID; ++k) acc += hid[g * HID + k] * Wl2[k * NC + c];
  raw[t] = acc;
}

// d_out layout: logits[256] | loss[1] | pred[128] | prob[256] | feature[16384]
__global__ void head_kernel(const float* __restrict__ raw,
                            const int* __restrict__ y,
                            float* __restrict__ out) {
  __shared__ float red[G_GRAPHS];
  int g = threadIdx.x;
  float r0 = raw[g * 2 + 0], r1 = raw[g * 2 + 1];
  float m  = fmaxf(r0, r1);
  float e0 = __expf(r0 - m), e1 = __expf(r1 - m);
  float Z  = e0 + e1;
  float lZ = __logf(Z);
  float l0 = (r0 - m) - lZ, l1 = (r1 - m) - lZ;
  out[g * 2 + 0] = l0;                       // logits
  out[g * 2 + 1] = l1;
  out[257 + g]   = (r1 > r0) ? 1.0f : 0.0f;  // pred (argmax, first-max ties -> 0)
  out[385 + g * 2 + 0] = e0 / Z;             // prob
  out[385 + g * 2 + 1] = e1 / Z;
  red[g] = -((y[g] == 0) ? l0 : l1);
  __syncthreads();
  for (int s2 = G_GRAPHS / 2; s2 > 0; s2 >>= 1) {
    if (g < s2) red[g] += red[g + s2];
    __syncthreads();
  }
  if (g == 0) out[256] = red[0] / (float)G_GRAPHS;   // loss
}

// ---------------- host side ----------------
static inline unsigned gridFor(long long n, int bs) {
  return (unsigned)((n + bs - 1) / bs);
}

static void launch_gemm(const float* A, const _Float16* Bp, float* C,
                        int M, int K, int Nc, const float* bias, hipStream_t s) {
  int tilesM2 = ((M >> 4) + 1) >> 1;
  int tilesN2 = ((Nc >> 4) + 1) >> 1;
  int waves = tilesM2 * tilesN2;
  const int wpb = 4;
  dim3 block(32 * wpb);
  dim3 grid((waves + wpb - 1) / wpb);
  if (bias)
    gemm_wmma_kernel<true><<<grid, block, 0, s>>>(A, Bp, C, M, K, Nc, bias);
  else
    gemm_wmma_kernel<false><<<grid, block, 0, s>>>(A, Bp, C, M, K, Nc, nullptr);
}

extern "C" void kernel_launch(void* const* d_in, const int* in_sizes, int n_in,
                              void* d_out, int out_size, void* d_ws, size_t ws_size,
                              hipStream_t stream) {
  (void)in_sizes; (void)n_in; (void)out_size; (void)ws_size;

  const float* x     = (const float*)d_in[0];
  const int*   ei    = (const int*)d_in[1];        // [2, E] flat
  const int*   batch = (const int*)d_in[2];
  const int*   y     = (const int*)d_in[3];
  const float* W1    = (const float*)d_in[4];
  const float* b1    = (const float*)d_in[5];
  const float* Wg1   = (const float*)d_in[6];
  const float* asrc1 = (const float*)d_in[7];
  const float* adst1 = (const float*)d_in[8];
  const float* bg1   = (const float*)d_in[9];
  const float* Wg2   = (const float*)d_in[10];
  const float* asrc2 = (const float*)d_in[11];
  const float* adst2 = (const float*)d_in[12];
  const float* bg2   = (const float*)d_in[13];
  const float* Wl1   = (const float*)d_in[14];
  const float* bl1   = (const float*)d_in[15];
  const float* Wl2   = (const float*)d_in[16];
  const float* bl2   = (const float*)d_in[17];

  const int* row = ei;            // edge_index[0]
  const int* col = ei + N_EDGES;  // edge_index[1]

  float* out = (float*)d_out;
  float* feature = out + 641;     // [G, HID] lives directly in d_out

  // workspace carve-up (floats; pack buffers 32B-aligned by construction)
  float* ws = (float*)d_ws;
  size_t o = 0;
  float* h1   = ws + o; o += (size_t)N_NODES * LAT;
  float* h2   = ws + o; o += (size_t)N_NODES * HL;
  float* h3   = ws + o; o += (size_t)N_NODES * HL;
  float* hW   = ws + o; o += (size_t)N_NODES * HL;   // current layer x@W
  float* accb = ws + o; o += (size_t)N_NODES * HL;   // scatter accumulator
  float* sbuf = ws + o; o += (size_t)N_NODES * H_HEADS;
  float* dbuf = ws + o; o += (size_t)N_NODES * H_HEADS;
  float* zbuf = ws + o; o += (size_t)N_NODES * H_HEADS; // also used as deg[N]
  float* catG = ws + o; o += (size_t)G_GRAPHS * CAT;
  float* hid  = ws + o; o += (size_t)G_GRAPHS * HID;
  float* raw  = ws + o; o += (size_t)G_GRAPHS * NC;
  int*   idxb = (int*)(ws + o); o += G_GRAPHS;
  _Float16* packs = (_Float16*)(ws + o);
  _Float16* W1p  = packs;                 // 128*32
  _Float16* Wg1p = W1p  + IN_DIM * LAT;   // 32*256
  _Float16* Wg2p = Wg1p + LAT * HL;       // 256*256
  _Float16* Wl1p = Wg2p + HL * HL;        // 544*128

  const int BS = 256;

  // ---------- pack weights into WMMA B-fragment order (f16) ----------
  pack_b_kernel<<<gridFor(IN_DIM * LAT, BS), BS, 0, stream>>>(W1,  W1p,  IN_DIM, LAT);
  pack_b_kernel<<<gridFor(LAT * HL,   BS), BS, 0, stream>>>(Wg1, Wg1p, LAT, HL);
  pack_b_kernel<<<gridFor(HL * HL,    BS), BS, 0, stream>>>(Wg2, Wg2p, HL, HL);
  pack_b_kernel<<<gridFor(CAT * HID,  BS), BS, 0, stream>>>(Wl1, Wl1p, CAT, HID);

  // ---------- GCN layer ----------
  launch_gemm(x, W1p, hW, N_NODES, IN_DIM, LAT, nullptr, stream);  // hW = x@W1
  fill_kernel<<<gridFor(N_NODES, BS), BS, 0, stream>>>(zbuf, 1.0f, N_NODES);  // deg=1 (self loop)
  gcn_deg_kernel<<<gridFor(N_EDGES, BS), BS, 0, stream>>>(col, zbuf);
  fill_kernel<<<gridFor((long long)N_NODES * LAT, BS), BS, 0, stream>>>(accb, 0.0f, (long long)N_NODES * LAT);
  gcn_scatter_kernel<<<gridFor((long long)N_EDGES_SL * LAT, BS), BS, 0, stream>>>(row, col, zbuf, hW, accb);
  bias_elu_kernel<<<gridFor((long long)N_NODES * LAT, BS), BS, 0, stream>>>(accb, b1, h1, LAT, (long long)N_NODES * LAT);

  // ---------- GAT layers ----------
  struct GatParams { const float* hin; int K; const _Float16* Wp; const float* as;
                     const float* ad; const float* bg; float* hout; };
  GatParams layers[2] = {
    { h1, LAT, Wg1p, asrc1, adst1, bg1, h2 },
    { h2, HL,  Wg2p, asrc2, adst2, bg2, h3 },
  };
  for (int L = 0; L < 2; ++L) {
    const GatParams& P = layers[L];
    launch_gemm(P.hin, P.Wp, hW, N_NODES, P.K, HL, nullptr, stream);  // hW = h@Wg
    gat_sd_kernel<<<gridFor(N_NODES * H_HEADS, BS), BS, 0, stream>>>(hW, P.as, P.ad, sbuf, dbuf);
    fill_kernel<<<gridFor((long long)N_NODES * H_HEADS, BS), BS, 0, stream>>>(zbuf, 0.0f, (long long)N_NODES * H_HEADS);
    gat_z_kernel<<<gridFor((long long)N_EDGES_SL * H_HEADS, BS), BS, 0, stream>>>(row, col, sbuf, dbuf, zbuf);
    fill_kernel<<<gridFor((long long)N_NODES * HL, BS), BS, 0, stream>>>(accb, 0.0f, (long long)N_NODES * HL);
    gat_scatter_kernel<<<gridFor((long long)N_EDGES_SL * HL, BS), BS, 0, stream>>>(row, col, sbuf, dbuf, zbuf, hW, accb);
    bias_elu_kernel<<<gridFor((long long)N_NODES * HL, BS), BS, 0, stream>>>(accb, P.bg, P.hout, HL, (long long)N_NODES * HL);
  }

  // ---------- readout ----------
  idx_kernel<<<1, G_GRAPHS, 0, stream>>>(batch, idxb);
  gather_cat_kernel<<<gridFor(G_GRAPHS * CAT, BS), BS, 0, stream>>>(idxb, h1, h2, h3, catG);
  launch_gemm(catG, Wl1p, feature, G_GRAPHS, CAT, HID, bl1, stream);  // feature -> d_out
  hidden_kernel<<<gridFor(G_GRAPHS * HID, BS), BS, 0, stream>>>(feature, hid);
  raw_kernel<<<gridFor(G_GRAPHS * NC, BS), BS, 0, stream>>>(hid, Wl2, bl2, raw);
  head_kernel<<<1, G_GRAPHS, 0, stream>>>(raw, y, out);
}
